// MultiHeadAttention_72335839199815
// MI455X (gfx1250) — matrix operand order repaired and checked by reading, not compiled, not run on hardware
//
#include <hip/hip_runtime.h>

// ---------------------------------------------------------------------------
// MI455X (gfx1250) fused multi-head attention, bf16 WMMA + f32 softmax.
// Bandwidth-bound problem (~260 MB mandatory HBM traffic @23.3 TB/s);
// attention matrix [B,S,S] is produced and consumed exactly once (fused).
// ---------------------------------------------------------------------------

#define HIDN   256
#define NHEADS 4
#define DHEAD  64
#define BB     64
#define SEQ    512
#define MROWS  (BB * SEQ)   // 32768

typedef __attribute__((ext_vector_type(16))) __bf16         v16bf;
typedef __attribute__((ext_vector_type(8)))  float          v8f;
typedef __attribute__((ext_vector_type(4)))  float          f32x4;
typedef __attribute__((ext_vector_type(8)))  unsigned short ushort8;

union BF16Tile {
    v16bf          v;
    unsigned short u[16];
    ushort8        h[2];
};

__device__ __forceinline__ unsigned short f2bf(float f) {
    unsigned int u = __float_as_uint(f);
    u += 0x7FFFu + ((u >> 16) & 1u);         // round-to-nearest-even
    return (unsigned short)(u >> 16);
}

__device__ __forceinline__ v8f wmma_bf16(const BF16Tile& a, const BF16Tile& b, v8f c) {
    // D = A(16x32 bf16) * B(32x16 bf16) + C(16x16 f32)
    return __builtin_amdgcn_wmma_f32_16x16x32_bf16(
        /*neg_a=*/false, a.v, /*neg_b=*/false, b.v,
        /*c_mod=*/(short)0, c, /*reuse_a=*/false, /*reuse_b=*/false);
}

// ---------------------------------------------------------------------------
// Weight convert + transpose: dst[n*K + k] = bf16(src[k*N + n])  (B-matrix
// sources must be column-major [N][K] so each lane reads two runs of 8 contiguous K vals)
// ---------------------------------------------------------------------------
__global__ void cvt_transpose_bf16(const float* __restrict__ src,
                                   unsigned short* __restrict__ dst,
                                   int K, int N) {
    int idx = blockIdx.x * blockDim.x + threadIdx.x;
    if (idx >= K * N) return;
    int k = idx / N, n = idx % N;
    dst[n * K + k] = f2bf(src[idx]);
}

// ---------------------------------------------------------------------------
// Generic WMMA GEMM:  D[M=32768, Ntot] = X[M,Kdim] * W[Kdim,Ntot] + bias.
// One wave computes a 16x64 tile (4 accumulators). Store modes:
//   0: bf16, scattered to [B, H, S, 64]   (Q / K projections)
//   1: bf16, transposed to [B, 64, S]     (V projection)
//   2: f32, row-major [M, Ntot]           (final output projection)
// ---------------------------------------------------------------------------
template <int MODE>
__global__ void __launch_bounds__(128)
proj_gemm(const float* __restrict__ X, const unsigned short* __restrict__ WT,
          const float* __restrict__ bias, int Kdim, int Ntot,
          void* __restrict__ outp) {
    const int lane = threadIdx.x & 31;
    const int wv   = threadIdx.x >> 5;
    const int gw   = blockIdx.x * 4 + wv;
    const int ntn  = Ntot >> 6;
    const int m0   = (gw / ntn) << 4;
    const int n0   = (gw % ntn) << 6;
    const int l15  = lane & 15;
    const int hi   = lane >> 4;            // 0 or 1
    const int row  = m0 + l15;             // A-matrix row for this lane

    v8f acc[4];
#pragma unroll
    for (int s = 0; s < 4; ++s) acc[s] = (v8f){0.f,0.f,0.f,0.f,0.f,0.f,0.f,0.f};

    for (int k0 = 0; k0 < Kdim; k0 += 32) {
        // A tile: lane holds K = {k0+8*hi .. +7} and {k0+16+8*hi .. +7}
        BF16Tile a;
        const float* ap = X + (size_t)row * Kdim + k0 + hi * 8;
        f32x4 c0 = *(const f32x4*)(ap);
        f32x4 c1 = *(const f32x4*)(ap + 4);
        f32x4 c2 = *(const f32x4*)(ap + 16);
        f32x4 c3 = *(const f32x4*)(ap + 20);
#pragma unroll
        for (int j = 0; j < 4; ++j) {
            a.u[j]      = f2bf(c0[j]);
            a.u[4 + j]  = f2bf(c1[j]);
            a.u[8 + j]  = f2bf(c2[j]);
            a.u[12 + j] = f2bf(c3[j]);
        }
#pragma unroll
        for (int s = 0; s < 4; ++s) {
            int col = n0 + s * 16 + l15;   // B-matrix column for this lane
            const unsigned short* bp = WT + (size_t)col * Kdim + k0 + hi * 8;
            BF16Tile bm;
            bm.h[0] = *(const ushort8*)(bp);
            bm.h[1] = *(const ushort8*)(bp + 16);
            acc[s] = wmma_bf16(a, bm, acc[s]);
        }
    }

    const int rbase = m0 + (hi << 3);
#pragma unroll
    for (int s = 0; s < 4; ++s) {
        int col = n0 + s * 16 + l15;
        float bv = bias[col];
#pragma unroll
        for (int r = 0; r < 8; ++r) {
            float val = acc[s][r] + bv;    // C/D layout: row = 8*hi + r, col = l15
            int orow = rbase + r;
            if (MODE == 2) {
                ((float*)outp)[(size_t)orow * Ntot + col] = val;
            } else {
                unsigned short bvv = f2bf(val);
                int b = orow >> 9, si = orow & 511;
                if (MODE == 0) {
                    int hh = col >> 6, dd = col & 63;
                    ((unsigned short*)outp)[((((b * 4 + hh) << 9) + si) << 6) + dd] = bvv;
                } else {
                    ((unsigned short*)outp)[(((b << 6) + col) << 9) + si] = bvv;
                }
            }
        }
    }
}

// ---------------------------------------------------------------------------
// Fused attention: energy (WMMA) -> mask/scale/softmax (f32, LDS) ->
// head-mean -> attention out + ctx = attn * V (WMMA).
// Block = 128 threads (4 waves), one (b, 16-row q-tile) per block.
// Dynamic LDS: E[16][512] f32 + A[16][512] f32 = 64 KB.
// ---------------------------------------------------------------------------
__global__ void __launch_bounds__(128)
attention_fused(const unsigned short* __restrict__ Qbf,
                const unsigned short* __restrict__ Kbf,
                const unsigned short* __restrict__ Vt,
                const int* __restrict__ mask,
                float* __restrict__ attn_out,
                float* __restrict__ ctx) {
    extern __shared__ float smem[];
    float* E = smem;              // 16*512 energy / probs scratch (per head)
    float* A = smem + 16 * 512;   // 16*512 head-mean attention accumulator

    const int b   = blockIdx.x >> 5;
    const int q0  = (blockIdx.x & 31) << 4;
    const int tid = threadIdx.x;
    const int lane = tid & 31, w = tid >> 5;
    const int l15 = lane & 15, hi = lane >> 4;

    for (int h = 0; h < NHEADS; ++h) {
        // ---- energy = Q * K^T : wave w handles n-tiles {w, w+4, ...} ----
        BF16Tile qa[2];
        {
            const unsigned short* qp =
                Qbf + (size_t)(((b * 4 + h) << 9) + q0 + l15) * DHEAD;
#pragma unroll
            for (int kk = 0; kk < 2; ++kk) {
                int off = kk * 32 + hi * 8;
                qa[kk].h[0] = *(const ushort8*)(qp + off);
                qa[kk].h[1] = *(const ushort8*)(qp + off + 16);
            }
        }
        const unsigned short* kbase = Kbf + ((size_t)(b * 4 + h) << 9) * DHEAD;
        for (int nt = w; nt < 32; nt += 4) {   // uniform per wave: EXEC all 1s
            int n0 = nt << 4;
            v8f acc = (v8f){0.f,0.f,0.f,0.f,0.f,0.f,0.f,0.f};
#pragma unroll
            for (int kk = 0; kk < 2; ++kk) {
                const unsigned short* kp =
                    kbase + (size_t)(n0 + l15) * DHEAD + kk * 32 + hi * 8;
                BF16Tile bm;
                bm.h[0] = *(const ushort8*)(kp);
                bm.h[1] = *(const ushort8*)(kp + 16);
                acc = wmma_bf16(qa[kk], bm, acc);
            }
#pragma unroll
            for (int r = 0; r < 8; ++r)
                E[(hi * 8 + r) * 512 + n0 + l15] = acc[r];
        }
        __syncthreads();

        // ---- scale, mask, softmax, head-mean accumulate ----
        {
            int rowi = tid >> 3, sub = tid & 7;     // 8 lanes per row
            const int* mrow = mask + ((size_t)(b * SEQ + q0 + rowi)) * SEQ;
            float* erow = E + rowi * 512;
            float mx = -3.0e38f;
            for (int c = sub; c < 512; c += 8) {
                float v = erow[c] * 0.125f;          // / sqrt(64)
                if (mrow[c] == 0) v = -1.0e10f;
                erow[c] = v;
                mx = fmaxf(mx, v);
            }
            mx = fmaxf(mx, __shfl_xor(mx, 1, 8));
            mx = fmaxf(mx, __shfl_xor(mx, 2, 8));
            mx = fmaxf(mx, __shfl_xor(mx, 4, 8));
            float sum = 0.f;
            for (int c = sub; c < 512; c += 8) {
                float e = __expf(erow[c] - mx);
                erow[c] = e;
                sum += e;
            }
            sum += __shfl_xor(sum, 1, 8);
            sum += __shfl_xor(sum, 2, 8);
            sum += __shfl_xor(sum, 4, 8);
            float sc = 0.25f / sum;                  // head-mean factor
            float* arow = A + rowi * 512;
            if (h == 0) { for (int c = sub; c < 512; c += 8) arow[c]  = erow[c] * sc; }
            else        { for (int c = sub; c < 512; c += 8) arow[c] += erow[c] * sc; }
        }
        __syncthreads();
    }

    // ---- write attention [B,S,S] output (rows q0..q0+15 are contiguous) ----
    {
        float* dst = attn_out + ((size_t)(b * SEQ + q0)) * SEQ;
        for (int i = tid; i < 16 * 512; i += 128) dst[i] = A[i];
    }

    // ---- ctx = attn * V : wave w computes d-columns [w*16, w*16+16) ----
    {
        int colc = w * 16 + l15;
        const unsigned short* vb = Vt + (((size_t)((b << 6) + colc)) << 9);
        v8f acc = (v8f){0.f,0.f,0.f,0.f,0.f,0.f,0.f,0.f};
        for (int k0 = 0; k0 < 512; k0 += 32) {
            BF16Tile am;
            const float* ap = A + l15 * 512 + k0 + hi * 8;
            f32x4 c0 = *(const f32x4*)(ap);
            f32x4 c1 = *(const f32x4*)(ap + 4);
            f32x4 c2 = *(const f32x4*)(ap + 16);
            f32x4 c3 = *(const f32x4*)(ap + 20);
#pragma unroll
            for (int j = 0; j < 4; ++j) {
                am.u[j]      = f2bf(c0[j]);
                am.u[4 + j]  = f2bf(c1[j]);
                am.u[8 + j]  = f2bf(c2[j]);
                am.u[12 + j] = f2bf(c3[j]);
            }
            BF16Tile bm;
            const unsigned short* bp = vb + k0 + hi * 8;
            bm.h[0] = *(const ushort8*)(bp);
            bm.h[1] = *(const ushort8*)(bp + 16);
            acc = wmma_bf16(am, bm, acc);
        }
        float* cp = ctx + (size_t)(b * SEQ + q0 + hi * 8) * DHEAD;
#pragma unroll
        for (int r = 0; r < 8; ++r) cp[r * DHEAD + colc] = acc[r];
    }
}

// ---------------------------------------------------------------------------
extern "C" void kernel_launch(void* const* d_in, const int* in_sizes, int n_in,
                              void* d_out, int out_size, void* d_ws, size_t ws_size,
                              hipStream_t stream) {
    (void)in_sizes; (void)n_in; (void)out_size; (void)ws_size;

    const float* query = (const float*)d_in[0];
    const float* key   = (const float*)d_in[1];
    const float* value = (const float*)d_in[2];
    const int*   mask  = (const int*)d_in[3];
    const float* Wq = (const float*)d_in[4];  const float* bq = (const float*)d_in[5];
    const float* Wk = (const float*)d_in[6];  const float* bk = (const float*)d_in[7];
    const float* Wv = (const float*)d_in[8];  const float* bv = (const float*)d_in[9];
    const float* Wh = (const float*)d_in[10]; const float* bh = (const float*)d_in[11];

    float* out      = (float*)d_out;                          // [B,S,256]
    float* attn_out = out + (size_t)BB * SEQ * HIDN;          // [B,S,S]

    // Workspace carve-up (~44 MB total)
    char* ws = (char*)d_ws;
    unsigned short* WqT = (unsigned short*)ws;  ws += 256 * 256 * 2;  // [N=256][K=256]
    unsigned short* WkT = (unsigned short*)ws;  ws += 256 * 256 * 2;
    unsigned short* WvT = (unsigned short*)ws;  ws += 64 * 256 * 2;   // [N=64][K=256]
    unsigned short* WhT = (unsigned short*)ws;  ws += 256 * 64 * 2;   // [N=256][K=64]
    unsigned short* Qbf = (unsigned short*)ws;  ws += (size_t)BB * NHEADS * SEQ * DHEAD * 2;
    unsigned short* Kbf = (unsigned short*)ws;  ws += (size_t)BB * NHEADS * SEQ * DHEAD * 2;
    unsigned short* Vt  = (unsigned short*)ws;  ws += (size_t)BB * DHEAD * SEQ * 2;
    float*          ctx = (float*)ws;           ws += (size_t)BB * SEQ * DHEAD * 4;

    // 1) weights -> bf16, column-major [N][K]
    cvt_transpose_bf16<<<(256 * 256 + 255) / 256, 256, 0, stream>>>(Wq, WqT, 256, 256);
    cvt_transpose_bf16<<<(256 * 256 + 255) / 256, 256, 0, stream>>>(Wk, WkT, 256, 256);
    cvt_transpose_bf16<<<(256 * 64 + 255) / 256, 256, 0, stream>>>(Wv, WvT, 256, 64);
    cvt_transpose_bf16<<<(64 * 256 + 255) / 256, 256, 0, stream>>>(Wh, WhT, 64, 256);

    // 2) projections (WMMA). Waves = (M/16)*(N/64); 4 waves/block.
    proj_gemm<0><<<(MROWS / 16) * (HIDN / 64) / 4, 128, 0, stream>>>(
        query, WqT, bq, 256, 256, Qbf);
    proj_gemm<0><<<(MROWS / 16) * (HIDN / 64) / 4, 128, 0, stream>>>(
        key, WkT, bk, 256, 256, Kbf);
    proj_gemm<1><<<(MROWS / 16) * (DHEAD / 64) / 4, 128, 0, stream>>>(
        value, WvT, bv, 256, 64, Vt);

    // 3) fused attention (64 KB dynamic LDS per block)
    attention_fused<<<BB * (SEQ / 16), 128, 2 * 16 * 512 * sizeof(float), stream>>>(
        Qbf, Kbf, Vt, mask, attn_out, ctx);

    // 4) output projection: [32768,64] x [64,256] + bh -> d_out
    proj_gemm<2><<<(MROWS / 16) * (HIDN / 64) / 4, 128, 0, stream>>>(
        ctx, WhT, bh, 64, 256, out);
}
